// CNN_LSTM_64939905515952
// MI455X (gfx1250) — matrix operand order repaired and checked by reading, not compiled, not run on hardware
//
#include <hip/hip_runtime.h>
#include <cstdint>
#include <cstddef>

#define BATCH 128
#define FEAT  128
#define HID   512
#define FOURH 2048
#define SEQ   512
#define PRED  128
#define TSTEPS (SEQ + PRED - 1)
#define NWG   64
#define K0TOT 640    /* FEAT + HID */
#define K1TOT 1024   /* HID + HID  */

typedef __attribute__((ext_vector_type(16))) _Float16 v16h;
typedef __attribute__((ext_vector_type(8)))  _Float16 v8h;
typedef __attribute__((ext_vector_type(8)))  float    v8f;
typedef __attribute__((ext_vector_type(4)))  int      v4i;

union Frag { v16h v; v8h h[2]; };

#if __has_builtin(__builtin_amdgcn_global_load_async_to_lds_b128) && \
    __has_builtin(__builtin_amdgcn_s_wait_asynccnt)
#define HAVE_ASYNC_LDS 1
#else
#define HAVE_ASYNC_LDS 0
#endif

// ---------------------------------------------------------------------------
// Fragment loaders matching CDNA5 WMMA VGPR layouts (ISA 7.12.2):
//  A 16x32 f16: lane m=lane&15 holds row m; lanes<16 hold K=kb..kb+7, 16+kb..;
//  B 32x16 f16: lane n=lane&15 holds column n; lanes<16 hold K=kb..kb+7, 16+kb.. (same as A).
//  Weights stored row-major [outCol x K] => B column n = contiguous weight row.
// ---------------------------------------------------------------------------
__device__ __forceinline__ v16h loadA(const _Float16* base, int ld, int lane) {
  int m  = lane & 15;
  int kb = (lane >> 4) << 3;                 // 0 or 8
  const _Float16* p = base + (size_t)m * ld + kb;
  Frag f;
  f.h[0] = *(const v8h*)(p);                 // K = kb .. kb+7
  f.h[1] = *(const v8h*)(p + 16);            // K = 16+kb .. 16+kb+7
  return f.v;
}

__device__ __forceinline__ v16h loadB(const _Float16* wbase, int ld, int lane) {
  int n  = lane & 15;
  int kb = (lane >> 4) << 3;                 // 0 or 8
  const _Float16* p = wbase + (size_t)n * ld + kb;
  Frag f;
  f.h[0] = *(const v8h*)(p);                 // K = kb .. kb+7
  f.h[1] = *(const v8h*)(p + 16);            // K = 16+kb .. 16+kb+7
  return f.v;
}

__device__ __forceinline__ v8f wmma16(v16h a, v16h b, v8f c) {
  return __builtin_amdgcn_wmma_f32_16x16x32_f16(false, a, false, b, (short)0, c,
                                                false, false);
}

__device__ __forceinline__ float sigf(float x) { return 1.0f / (1.0f + __expf(-x)); }

// Sense-reversal grid barrier across NWG persistent workgroups.
__device__ __forceinline__ void grid_barrier(unsigned* cnt, unsigned* gen) {
  __threadfence();
  __syncthreads();
  if (threadIdx.x == 0) {
    unsigned g = __atomic_load_n(gen, __ATOMIC_ACQUIRE);
    unsigned a = __atomic_add_fetch(cnt, 1u, __ATOMIC_ACQ_REL);
    if (a == (unsigned)NWG) {
      __atomic_store_n(cnt, 0u, __ATOMIC_RELAXED);
      __atomic_fetch_add(gen, 1u, __ATOMIC_ACQ_REL);
    } else {
      while (__atomic_load_n(gen, __ATOMIC_ACQUIRE) == g)
        __builtin_amdgcn_s_sleep(1);
    }
  }
  __syncthreads();
}

// Encoder: e = inp @ enc_w.T + enc_b for one WG (16 output cols, 8 row tiles).
template <bool FROMX>
__device__ __forceinline__ void encoder_step(int t, const float* __restrict__ x,
                                             const _Float16* __restrict__ OUTS,
                                             const _Float16* __restrict__ ENCW,
                                             const float* __restrict__ ENCB,
                                             _Float16* __restrict__ EB,
                                             int wg, int wave, int lane) {
  const int nc0  = wg * 16;
  const int rowt = wave;
  v8f acc = {};
  for (int kt = 0; kt < FEAT / 32; ++kt) {
    int k0 = kt * 32;
    v16h a;
    if (FROMX) {
      int m = lane & 15, kb = (lane >> 4) << 3;
      const float* xp = x + (size_t)(rowt * 16 + m) * (SEQ * FEAT)
                          + (size_t)t * FEAT + k0 + kb;
      __builtin_prefetch(xp + FEAT, 0, 1);     // next timestep column
      Frag f;
#pragma unroll
      for (int e = 0; e < 8; ++e) {
        f.h[0][e] = (_Float16)xp[e];
        f.h[1][e] = (_Float16)xp[16 + e];
      }
      a = f.v;
    } else {
      a = loadA(OUTS + rowt * 16 * FEAT + k0, FEAT, lane);   // autoregressive
    }
    v16h b = loadB(ENCW + (size_t)nc0 * FEAT + k0, FEAT, lane);
    acc = wmma16(a, b, acc);
  }
  int n = lane & 15, roff = (lane >> 4) * 8;
  float bias = ENCB[nc0 + n];
#pragma unroll
  for (int v = 0; v < 8; ++v) {
    int r = rowt * 16 + roff + v;
    EB[(size_t)r * FEAT + nc0 + n] = (_Float16)(acc[v] + bias);
  }
}

// ---------------------------------------------------------------------------
// Prologue: pack weights f32 -> f16 into per-WG LDS-source blocks, combine
// biases, zero h-state / scratch / barrier counters. Runs every launch.
// Gate row mapping for WG w, local row r (0..31):
//   gr = (r>>3)*HID + w*8 + (r&7)   (i rows 0..7, f 8..15, g 16..23, o 24..31)
// ---------------------------------------------------------------------------
__global__ void pack_kernel(const float* __restrict__ enc_w, const float* __restrict__ enc_b,
                            const float* __restrict__ dec_w, const float* __restrict__ dec_b,
                            const float* __restrict__ wih0,  const float* __restrict__ whh0,
                            const float* __restrict__ bih0,  const float* __restrict__ bhh0,
                            const float* __restrict__ wih1,  const float* __restrict__ whh1,
                            const float* __restrict__ bih1,  const float* __restrict__ bhh1,
                            _Float16* __restrict__ W0, _Float16* __restrict__ W1,
                            _Float16* __restrict__ ENCW, _Float16* __restrict__ DECW,
                            float* __restrict__ B0, float* __restrict__ B1,
                            float* __restrict__ ENCB, float* __restrict__ DECB,
                            _Float16* __restrict__ H0, _Float16* __restrict__ H1,
                            _Float16* __restrict__ OUTS, unsigned* __restrict__ CNT)
{
  const long stride = (long)gridDim.x * blockDim.x;
  const long t0 = (long)blockIdx.x * blockDim.x + threadIdx.x;

  for (long i = t0; i < (long)NWG * 32 * K0TOT; i += stride) {
    int col = (int)(i % K0TOT); long rr = i / K0TOT;
    int r = (int)(rr & 31); int w = (int)(rr >> 5);
    int gr = (r >> 3) * HID + w * 8 + (r & 7);
    float v = (col < FEAT) ? wih0[(size_t)gr * FEAT + col]
                           : whh0[(size_t)gr * HID + (col - FEAT)];
    W0[i] = (_Float16)v;
  }
  for (long i = t0; i < (long)NWG * 32 * K1TOT; i += stride) {
    int col = (int)(i % K1TOT); long rr = i / K1TOT;
    int r = (int)(rr & 31); int w = (int)(rr >> 5);
    int gr = (r >> 3) * HID + w * 8 + (r & 7);
    float v = (col < HID) ? wih1[(size_t)gr * HID + col]
                          : whh1[(size_t)gr * HID + (col - HID)];
    W1[i] = (_Float16)v;
  }
  for (long i = t0; i < (long)FEAT * FEAT; i += stride) ENCW[i] = (_Float16)enc_w[i];
  for (long i = t0; i < (long)FEAT * HID;  i += stride) DECW[i] = (_Float16)dec_w[i];
  for (long i = t0; i < FOURH; i += stride) {
    int w = (int)(i >> 5), r = (int)(i & 31);
    int gr = (r >> 3) * HID + w * 8 + (r & 7);
    B0[i] = bih0[gr] + bhh0[gr];
    B1[i] = bih1[gr] + bhh1[gr];
  }
  for (long i = t0; i < FEAT; i += stride) { ENCB[i] = enc_b[i]; DECB[i] = dec_b[i]; }
  for (long i = t0; i < 2L * BATCH * HID; i += stride) {
    H0[i] = (_Float16)0.f; H1[i] = (_Float16)0.f;
  }
  for (long i = t0; i < (long)BATCH * FEAT; i += stride) OUTS[i] = (_Float16)0.f;
  if (t0 < 8) CNT[t0] = 0u;
}

// ---------------------------------------------------------------------------
// Persistent LSTM kernel: 64 WGs x 256 threads (8 wave32), weights LDS-resident.
// ---------------------------------------------------------------------------
__global__ __launch_bounds__(256, 1)
void lstm_persist(const float* __restrict__ x,
                  const _Float16* __restrict__ W0g, const _Float16* __restrict__ W1g,
                  const _Float16* __restrict__ ENCW, const _Float16* __restrict__ DECW,
                  const float* __restrict__ B0g, const float* __restrict__ B1g,
                  const float* __restrict__ ENCB, const float* __restrict__ DECB,
                  _Float16* __restrict__ H0, _Float16* __restrict__ H1,
                  _Float16* __restrict__ EB, _Float16* __restrict__ OUTS,
                  unsigned* __restrict__ CNT, float* __restrict__ out)
{
  extern __shared__ char smem[];
  _Float16* lw0 = (_Float16*)smem;                               // 32 x 640 f16
  _Float16* lw1 = (_Float16*)(smem + 32 * K0TOT * 2);            // 32 x 1024 f16
  float* gat = (float*)(smem + 32 * K0TOT * 2 + 32 * K1TOT * 2); // 128 x 32 f32
  float* c0s = gat + 128 * 32;                                   // 128 x 8 f32
  float* c1s = c0s + 128 * 8;                                    // 128 x 8 f32
  float* lb0 = c1s + 128 * 8;                                    // 32 f32
  float* lb1 = lb0 + 32;                                         // 32 f32

  const int wg   = blockIdx.x;
  const int tid  = threadIdx.x;
  const int wave = tid >> 5;
  const int lane = tid & 31;
  unsigned* cnt = CNT;
  unsigned* gen = CNT + 1;

  // ---- Stage this WG's weight slices to LDS once ---------------------------
#if HAVE_ASYNC_LDS
  {
    typedef __attribute__((address_space(1))) v4i* gvec_t;
    typedef __attribute__((address_space(3))) v4i* lvec_t;
    gvec_t s0 = (gvec_t)(W0g + (size_t)wg * 32 * K0TOT);
    lvec_t d0 = (lvec_t)(void*)lw0;
    for (int i = tid; i < (32 * K0TOT * 2) / 16; i += 256)
      __builtin_amdgcn_global_load_async_to_lds_b128(s0 + i, d0 + i, 0, 0);
    gvec_t s1 = (gvec_t)(W1g + (size_t)wg * 32 * K1TOT);
    lvec_t d1 = (lvec_t)(void*)lw1;
    for (int i = tid; i < (32 * K1TOT * 2) / 16; i += 256)
      __builtin_amdgcn_global_load_async_to_lds_b128(s1 + i, d1 + i, 0, 0);
    if (tid < 32) { lb0[tid] = B0g[wg * 32 + tid]; lb1[tid] = B1g[wg * 32 + tid]; }
    for (int i = tid; i < 128 * 8; i += 256) { c0s[i] = 0.f; c1s[i] = 0.f; }
    __builtin_amdgcn_s_wait_asynccnt(0);
  }
#else
  {
    const uint4* s0 = (const uint4*)(W0g + (size_t)wg * 32 * K0TOT);
    uint4* d0 = (uint4*)lw0;
    for (int i = tid; i < (32 * K0TOT * 2) / 16; i += 256) d0[i] = s0[i];
    const uint4* s1 = (const uint4*)(W1g + (size_t)wg * 32 * K1TOT);
    uint4* d1 = (uint4*)lw1;
    for (int i = tid; i < (32 * K1TOT * 2) / 16; i += 256) d1[i] = s1[i];
    if (tid < 32) { lb0[tid] = B0g[wg * 32 + tid]; lb1[tid] = B1g[wg * 32 + tid]; }
    for (int i = tid; i < 128 * 8; i += 256) { c0s[i] = 0.f; c1s[i] = 0.f; }
  }
#endif
  __syncthreads();

  // Encoder for t = 0 (always teacher-forced), then enter the steady loop.
  if (wg < 8)
    encoder_step<true>(0, x, OUTS, ENCW, ENCB, EB, wg, wave, lane);
  grid_barrier(cnt, gen);

  for (int t = 0; t < TSTEPS; ++t) {
    const _Float16* h0cur = H0 + (size_t)(t & 1) * BATCH * HID;
    _Float16*       h0nxt = H0 + (size_t)((t + 1) & 1) * BATCH * HID;
    const _Float16* h1cur = H1 + (size_t)(t & 1) * BATCH * HID;
    _Float16*       h1nxt = H1 + (size_t)((t + 1) & 1) * BATCH * HID;

    // ---- layer-0 gates + cell update (all WGs) -----------------------------
    {
      const int rowt = wave;
      v8f a0 = {}, a1 = {};
      const _Float16* ebase = EB + rowt * 16 * FEAT;
      for (int kt = 0; kt < FEAT / 32; ++kt) {
        int k0 = kt * 32;
        v16h a  = loadA(ebase + k0, FEAT, lane);
        v16h b0 = loadB(lw0 + k0, K0TOT, lane);               // gate rows 0..15
        v16h b1 = loadB(lw0 + 16 * K0TOT + k0, K0TOT, lane);  // gate rows 16..31
        a0 = wmma16(a, b0, a0);
        a1 = wmma16(a, b1, a1);
      }
      const _Float16* hbase = h0cur + (size_t)rowt * 16 * HID;
      for (int kt = 0; kt < HID / 32; ++kt) {
        int k0 = kt * 32;
        v16h a  = loadA(hbase + k0, HID, lane);
        v16h b0 = loadB(lw0 + FEAT + k0, K0TOT, lane);
        v16h b1 = loadB(lw0 + 16 * K0TOT + FEAT + k0, K0TOT, lane);
        a0 = wmma16(a, b0, a0);
        a1 = wmma16(a, b1, a1);
      }
      int n = lane & 15, roff = (lane >> 4) * 8;
#pragma unroll
      for (int v = 0; v < 8; ++v) {
        int r = rowt * 16 + roff + v;
        gat[r * 32 + n]      = a0[v] + lb0[n];
        gat[r * 32 + 16 + n] = a1[v] + lb0[16 + n];
      }
      __syncthreads();
#pragma unroll
      for (int i = 0; i < 4; ++i) {
        int idx = tid * 4 + i;
        int r = idx >> 3, cc = idx & 7;
        float gi = gat[r * 32 + cc],      gf = gat[r * 32 + 8 + cc];
        float gg = gat[r * 32 + 16 + cc], go = gat[r * 32 + 24 + cc];
        float c = sigf(gf) * c0s[idx] + sigf(gi) * tanhf(gg);
        c0s[idx] = c;
        h0nxt[(size_t)r * HID + wg * 8 + cc] = (_Float16)(sigf(go) * tanhf(c));
      }
    }
    grid_barrier(cnt, gen);

    // ---- layer-1 gates + cell update (all WGs) -----------------------------
    {
      const int rowt = wave;
      v8f a0 = {}, a1 = {};
      const _Float16* hbase0 = h0nxt + (size_t)rowt * 16 * HID;
      for (int kt = 0; kt < HID / 32; ++kt) {
        int k0 = kt * 32;
        v16h a  = loadA(hbase0 + k0, HID, lane);
        v16h b0 = loadB(lw1 + k0, K1TOT, lane);
        v16h b1 = loadB(lw1 + 16 * K1TOT + k0, K1TOT, lane);
        a0 = wmma16(a, b0, a0);
        a1 = wmma16(a, b1, a1);
      }
      const _Float16* hbase1 = h1cur + (size_t)rowt * 16 * HID;
      for (int kt = 0; kt < HID / 32; ++kt) {
        int k0 = kt * 32;
        v16h a  = loadA(hbase1 + k0, HID, lane);
        v16h b0 = loadB(lw1 + HID + k0, K1TOT, lane);
        v16h b1 = loadB(lw1 + 16 * K1TOT + HID + k0, K1TOT, lane);
        a0 = wmma16(a, b0, a0);
        a1 = wmma16(a, b1, a1);
      }
      int n = lane & 15, roff = (lane >> 4) * 8;
#pragma unroll
      for (int v = 0; v < 8; ++v) {
        int r = rowt * 16 + roff + v;
        gat[r * 32 + n]      = a0[v] + lb1[n];
        gat[r * 32 + 16 + n] = a1[v] + lb1[16 + n];
      }
      __syncthreads();
#pragma unroll
      for (int i = 0; i < 4; ++i) {
        int idx = tid * 4 + i;
        int r = idx >> 3, cc = idx & 7;
        float gi = gat[r * 32 + cc],      gf = gat[r * 32 + 8 + cc];
        float gg = gat[r * 32 + 16 + cc], go = gat[r * 32 + 24 + cc];
        float c = sigf(gf) * c1s[idx] + sigf(gi) * tanhf(gg);
        c1s[idx] = c;
        h1nxt[(size_t)r * HID + wg * 8 + cc] = (_Float16)(sigf(go) * tanhf(c));
      }
    }
    grid_barrier(cnt, gen);

    // ---- decoder (WGs 8..15) overlapped with next encoder when teacher-forced
    if (wg >= 8 && wg < 16) {
      const int dw = wg - 8;
      const int nc0 = dw * 16;
      const int rowt = wave;
      v8f acc = {};
      const _Float16* hbase = h1nxt + (size_t)rowt * 16 * HID;
      for (int kt = 0; kt < HID / 32; ++kt) {
        int k0 = kt * 32;
        v16h a = loadA(hbase + k0, HID, lane);
        v16h b = loadB(DECW + (size_t)nc0 * HID + k0, HID, lane);
        acc = wmma16(a, b, acc);
      }
      int n = lane & 15, roff = (lane >> 4) * 8;
      float bias = DECB[nc0 + n];
#pragma unroll
      for (int v = 0; v < 8; ++v) {
        int r = rowt * 16 + roff + v;
        float o = acc[v] + bias;
        OUTS[(size_t)r * FEAT + nc0 + n] = (_Float16)o;   // next-step input
        if (t >= SEQ - 1)
          out[(size_t)r * PRED * FEAT + (size_t)(t - (SEQ - 1)) * FEAT + nc0 + n] = o;
      }
    }
    if (wg < 8 && (t + 1) < SEQ)   // teacher-forced: independent of decoder
      encoder_step<true>(t + 1, x, OUTS, ENCW, ENCB, EB, wg, wave, lane);
    grid_barrier(cnt, gen);

    // Autoregressive tail: encoder needs OUTS, so it runs after the barrier.
    if ((t + 1) >= SEQ && (t + 1) < TSTEPS) {
      if (wg < 8)
        encoder_step<false>(t + 1, x, OUTS, ENCW, ENCB, EB, wg, wave, lane);
      grid_barrier(cnt, gen);
    }
  }
}

// ---------------------------------------------------------------------------
extern "C" void kernel_launch(void* const* d_in, const int* in_sizes, int n_in,
                              void* d_out, int out_size, void* d_ws, size_t ws_size,
                              hipStream_t stream)
{
  (void)in_sizes; (void)n_in; (void)out_size; (void)ws_size;
  const float* x     = (const float*)d_in[0];
  const float* enc_w = (const float*)d_in[1];
  const float* enc_b = (const float*)d_in[2];
  const float* dec_w = (const float*)d_in[3];
  const float* dec_b = (const float*)d_in[4];
  const float* wih0  = (const float*)d_in[5];
  const float* whh0  = (const float*)d_in[6];
  const float* bih0  = (const float*)d_in[7];
  const float* bhh0  = (const float*)d_in[8];
  const float* wih1  = (const float*)d_in[9];
  const float* whh1  = (const float*)d_in[10];
  const float* bih1  = (const float*)d_in[11];
  const float* bhh1  = (const float*)d_in[12];

  char* ws = (char*)d_ws;
  size_t off = 0;
  auto carve = [&](size_t bytes) -> char* {
    char* p = ws + off;
    off += (bytes + 255) & ~(size_t)255;
    return p;
  };
  _Float16* W0   = (_Float16*)carve((size_t)NWG * 32 * K0TOT * 2);
  _Float16* W1   = (_Float16*)carve((size_t)NWG * 32 * K1TOT * 2);
  _Float16* ENCW = (_Float16*)carve((size_t)FEAT * FEAT * 2);
  _Float16* DECW = (_Float16*)carve((size_t)FEAT * HID * 2);
  float*    B0   = (float*)carve(FOURH * 4);
  float*    B1   = (float*)carve(FOURH * 4);
  float*    ENCB = (float*)carve(FEAT * 4);
  float*    DECB = (float*)carve(FEAT * 4);
  _Float16* H0   = (_Float16*)carve((size_t)2 * BATCH * HID * 2);
  _Float16* H1   = (_Float16*)carve((size_t)2 * BATCH * HID * 2);
  _Float16* EB   = (_Float16*)carve((size_t)BATCH * FEAT * 2);
  _Float16* OUTS = (_Float16*)carve((size_t)BATCH * FEAT * 2);
  unsigned* CNT  = (unsigned*)carve(256);

  pack_kernel<<<512, 256, 0, stream>>>(enc_w, enc_b, dec_w, dec_b,
                                       wih0, whh0, bih0, bhh0,
                                       wih1, whh1, bih1, bhh1,
                                       W0, W1, ENCW, DECW, B0, B1, ENCB, DECB,
                                       H0, H1, OUTS, CNT);

  const int kSmem = 32 * K0TOT * 2 + 32 * K1TOT * 2   // weight slices
                  + 128 * 32 * 4                       // gate scratch
                  + 2 * 128 * 8 * 4                    // c0, c1
                  + 64 * 4;                            // biases
  (void)hipFuncSetAttribute((const void*)lstm_persist,
                            hipFuncAttributeMaxDynamicSharedMemorySize, kSmem);
  lstm_persist<<<NWG, 256, kSmem, stream>>>(x, W0, W1, ENCW, DECW,
                                            B0, B1, ENCB, DECB,
                                            H0, H1, EB, OUTS, CNT, (float*)d_out);
}
